// MultiHeadAttention_2568390443580
// MI455X (gfx1250) — matrix operand order repaired and checked by reading, not compiled, not run on hardware
//
#include <hip/hip_runtime.h>
#include <stdint.h>

#define D_MODEL 2048
#define N_HEADS 16
#define HEAD_DIM 128
#define BATCH 4
#define SEQ 2048
#define M_TOTAL (BATCH * SEQ)   /* 8192 */
#define KDIM    D_MODEL         /* 2048 */
#define N_QKV   (3 * D_MODEL)   /* 6144 */

typedef __attribute__((ext_vector_type(16))) __bf16       v16bf;
typedef __attribute__((ext_vector_type(8)))  float        v8f;
typedef __attribute__((ext_vector_type(4)))  unsigned int uint4v;
typedef __attribute__((ext_vector_type(4)))  int          int4v;

union FragB { v16bf v; uint4v u[2]; };

#if __has_builtin(__builtin_amdgcn_global_load_async_to_lds_b128)
#define USE_ASYNC 1
#else
#define USE_ASYNC 0
#endif

__device__ __forceinline__ unsigned short f32_to_bf16(float f) {
  union { float f; unsigned int u; } x; x.f = f;
  unsigned int r = x.u + 0x7FFFu + ((x.u >> 16) & 1u);   // round-to-nearest-even
  return (unsigned short)(r >> 16);
}

// 16-byte global -> LDS copy: async (ASYNCcnt) on CDNA5, sync fallback otherwise
__device__ __forceinline__ void cp16_g2s(const unsigned short* g, unsigned short* l) {
#if USE_ASYNC
  __builtin_amdgcn_global_load_async_to_lds_b128((int4v*)g, (int4v*)l, 0, 0);
#else
  *(uint4v*)l = *(const uint4v*)g;
#endif
}

__device__ __forceinline__ void wait_async_all() {
#if USE_ASYNC
#if __has_builtin(__builtin_amdgcn_s_wait_asynccnt)
  __builtin_amdgcn_s_wait_asynccnt(0);
#else
  asm volatile("s_wait_asynccnt 0" ::: "memory");
#endif
#endif
}

// ---------------------------------------------------------------------------
// fp32 -> bf16 elementwise (vectorized)
// ---------------------------------------------------------------------------
__global__ void cvt_bf16(const float* __restrict__ in,
                         unsigned short* __restrict__ out, int n) {
  int i = (blockIdx.x * 256 + threadIdx.x) * 4;
  if (i + 3 < n) {
    float4 v = *(const float4*)(in + i);
    ushort4 r;
    r.x = f32_to_bf16(v.x); r.y = f32_to_bf16(v.y);
    r.z = f32_to_bf16(v.z); r.w = f32_to_bf16(v.w);
    *(ushort4*)(out + i) = r;
  }
}

// ---------------------------------------------------------------------------
// fp32 [K][N] -> bf16 transposed [N][K], tiled through LDS
// ---------------------------------------------------------------------------
__global__ void cvt_transpose_bf16(const float* __restrict__ in,
                                   unsigned short* __restrict__ out,
                                   int K, int N) {
  __shared__ float tile[32][33];
  int n0 = blockIdx.x * 32, k0 = blockIdx.y * 32;
  int tx = threadIdx.x, ty = threadIdx.y;           // 32 x 8
  for (int i = 0; i < 32; i += 8)
    tile[ty + i][tx] = in[(size_t)(k0 + ty + i) * N + n0 + tx];
  __syncthreads();
  for (int i = 0; i < 32; i += 8)
    out[(size_t)(n0 + ty + i) * K + k0 + tx] = f32_to_bf16(tile[tx][ty + i]);
}

// ---------------------------------------------------------------------------
// bf16 GEMM: A[M][K] x Bt[N][K]^T -> 128x128 block tile, 8 waves, wave = 64x32
// mode 0: QKV epilogue (bias, Q scale, scatter q/k head-major, v transposed)
// mode 1: fp32 out epilogue (bias)
// ---------------------------------------------------------------------------
__global__ __launch_bounds__(256) void gemm_bf16(
    const unsigned short* __restrict__ A,
    const unsigned short* __restrict__ Bt,
    const float* __restrict__ bias,
    int mode,
    unsigned short* __restrict__ qb,
    unsigned short* __restrict__ kb,
    unsigned short* __restrict__ vtb,
    float* __restrict__ outf) {
  const int K = KDIM;
  int lane  = threadIdx.x & 31;
  int wave  = threadIdx.x >> 5;
  int mbase = blockIdx.y * 128 + (wave & 1) * 64;
  int nbase = blockIdx.x * 128 + (wave >> 1) * 32;
  int l15   = lane & 15;
  int aoff  = (lane < 16) ? 0 : 8;    // A-frag K split: {0-7,16-23} vs {8-15,24-31}
  int boff  = (lane < 16) ? 0 : 8;    // B-frag K split: {0-7,16-23} vs {8-15,24-31}

  v8f acc[4][2];
  for (int mt = 0; mt < 4; mt++)
    for (int nt = 0; nt < 2; nt++)
      for (int e = 0; e < 8; e++) acc[mt][nt][e] = 0.0f;

  for (int k0 = 0; k0 < K; k0 += 32) {
    FragB a[4], b[2];
    for (int mt = 0; mt < 4; mt++) {
      const unsigned short* p =
          A + (size_t)(mbase + mt * 16 + l15) * K + k0 + aoff;
      a[mt].u[0] = *(const uint4v*)p;
      a[mt].u[1] = *(const uint4v*)(p + 16);
    }
    for (int nt = 0; nt < 2; nt++) {
      const unsigned short* p =
          Bt + (size_t)(nbase + nt * 16 + l15) * K + k0 + boff;
      b[nt].u[0] = *(const uint4v*)p;
      b[nt].u[1] = *(const uint4v*)(p + 16);
    }
    for (int mt = 0; mt < 4; mt++)
      for (int nt = 0; nt < 2; nt++)
        acc[mt][nt] = __builtin_amdgcn_wmma_f32_16x16x32_bf16(
            false, a[mt].v, false, b[nt].v, (short)0, acc[mt][nt], false, false);
  }

  const float SCALE = 0.08838834764831843f;  // 1/sqrt(128)
  for (int mt = 0; mt < 4; mt++)
    for (int nt = 0; nt < 2; nt++) {
      int ncol = nbase + nt * 16 + l15;
      float bv = bias[ncol];
      for (int r = 0; r < 8; r++) {
        int m = mbase + mt * 16 + r + ((lane < 16) ? 0 : 8);
        float val = acc[mt][nt][r] + bv;
        if (mode == 0) {
          int bb = m >> 11, t = m & 2047;       // /SEQ, %SEQ
          int sec = ncol >> 11;                 // 0=Q,1=K,2=V
          int c = ncol & 2047;
          int h = c >> 7, d = c & 127;
          size_t bh = (size_t)(bb * N_HEADS + h);
          if (sec == 0)
            qb[(bh * SEQ + t) * HEAD_DIM + d] = f32_to_bf16(val * SCALE);
          else if (sec == 1)
            kb[(bh * SEQ + t) * HEAD_DIM + d] = f32_to_bf16(val);
          else
            vtb[(bh * HEAD_DIM + d) * SEQ + t] = f32_to_bf16(val);
        } else {
          outf[(size_t)m * D_MODEL + ncol] = val;
        }
      }
    }
}

// ---------------------------------------------------------------------------
// Flash attention with double-buffered async K/V staging in LDS.
// One wave = 16 query rows; block = 128 rows of one (b,h); key tiles of 64.
// q,k: [B,H,T,hd] bf16 (q pre-scaled); vt: [B,H,hd,T] bf16; ctx: [B,T,C] bf16
// ---------------------------------------------------------------------------
__global__ __launch_bounds__(256) void flash_attn(
    const unsigned short* __restrict__ q,
    const unsigned short* __restrict__ k,
    const unsigned short* __restrict__ vt,
    unsigned short* __restrict__ ctx) {
  __shared__ __align__(16) unsigned short Kt[2][64][136];  // keys x d   (+8 pad)
  __shared__ __align__(16) unsigned short Vt[2][128][72];  // d    x keys(+8 pad)
  __shared__ __align__(16) unsigned short P[8][16][72];    // per-wave P strip

  int tid  = threadIdx.x;
  int lane = tid & 31;
  int wave = tid >> 5;
  int l15  = lane & 15;
  int bh   = blockIdx.y;
  int bb   = bh >> 4, h = bh & 15;
  int qblk = blockIdx.x * 128;
  int q0   = qblk + wave * 16;

  const unsigned short* qb = q  + (size_t)bh * SEQ * HEAD_DIM;
  const unsigned short* kb = k  + (size_t)bh * SEQ * HEAD_DIM;
  const unsigned short* vb = vt + (size_t)bh * HEAD_DIM * SEQ;

  int aoff = (lane < 16) ? 0 : 8;
  int boff = (lane < 16) ? 0 : 8;

  // Q fragments (K-dim = 128 -> 4 chunks of 32), rows q0..q0+15, read once
  FragB qa[4];
  {
    const unsigned short* p = qb + (size_t)(q0 + l15) * HEAD_DIM;
    for (int kk = 0; kk < 4; kk++) {
      qa[kk].u[0] = *(const uint4v*)(p + kk * 32 + aoff);
      qa[kk].u[1] = *(const uint4v*)(p + kk * 32 + aoff + 16);
    }
  }

  // cooperative staging of one 64-key K/V tile into LDS buffer c
  auto stage = [&](int kt2, int c) {
    int tk2 = kt2 * 64;
    {
      int key = tid >> 2;                    // 4 threads per key row
      int dc  = (tid & 3) * 32;              // 64B sub-chunk of the 256B row
      const unsigned short* src = kb + (size_t)(tk2 + key) * HEAD_DIM + dc;
      unsigned short* dst = &Kt[c][key][dc];
      for (int i = 0; i < 4; i++) cp16_g2s(src + i * 8, dst + i * 8);
    }
    {
      int d  = tid >> 1;                     // 2 threads per d row
      int tc = (tid & 1) * 32;               // 64B sub-chunk of the 128B row
      const unsigned short* src = vb + (size_t)d * SEQ + tk2 + tc;
      unsigned short* dst = &Vt[c][d][tc];
      for (int i = 0; i < 4; i++) cp16_g2s(src + i * 8, dst + i * 8);
    }
  };

  float mrow[8], lrow[8];
  for (int r = 0; r < 8; r++) { mrow[r] = -1e30f; lrow[r] = 0.0f; }
  v8f o[8];
  for (int jd = 0; jd < 8; jd++)
    for (int e = 0; e < 8; e++) o[jd][e] = 0.0f;

  int ktiles_wave = (q0 + 79) >> 6;          // causal bound for this wave
  int ktiles_blk  = (qblk + 191) >> 6;       // max over waves in block

  stage(0, 0);
  for (int kt = 0; kt < ktiles_blk; kt++) {
    int cur = kt & 1;
    wait_async_all();                        // this wave's copies into buf cur done
    __syncthreads();                         // everyone's copies done; prev reads done
    if (kt + 1 < ktiles_blk) stage(kt + 1, 1 - cur);

    if (kt < ktiles_wave) {                  // wave-uniform: EXEC all-ones inside
      int tk = kt * 64;

      // S = Q K^T (4 column tiles of 16 keys, K-dim 128), K-frags from LDS
      v8f s[4];
      for (int j = 0; j < 4; j++)
        for (int e = 0; e < 8; e++) s[j][e] = 0.0f;
      for (int j = 0; j < 4; j++) {
        const unsigned short* pk = &Kt[cur][j * 16 + l15][0];
        for (int kk = 0; kk < 4; kk++) {
          FragB bf;
          bf.u[0] = *(const uint4v*)(pk + kk * 32 + boff);
          bf.u[1] = *(const uint4v*)(pk + kk * 32 + boff + 16);
          s[j] = __builtin_amdgcn_wmma_f32_16x16x32_bf16(
              false, qa[kk].v, false, bf.v, (short)0, s[j], false, false);
        }
      }

      // causal mask (C layout: elem r -> row q0+r(+8), lane -> key column)
      for (int j = 0; j < 4; j++) {
        int key = tk + j * 16 + l15;
        if (key > q0) {
          for (int r = 0; r < 8; r++) {
            int qrow = q0 + r + ((lane < 16) ? 0 : 8);
            if (key > qrow) s[j][r] = -1e30f;
          }
        }
      }

      // online softmax: row max / rescale / row sum (xor-reduce in 16-lane half)
      float alpha[8];
      for (int r = 0; r < 8; r++) {
        float mx = fmaxf(fmaxf(s[0][r], s[1][r]), fmaxf(s[2][r], s[3][r]));
        for (int off = 1; off < 16; off <<= 1)
          mx = fmaxf(mx, __shfl_xor(mx, off, 32));
        float mnew = fmaxf(mrow[r], mx);
        alpha[r] = __expf(mrow[r] - mnew);
        mrow[r]  = mnew;
      }
      for (int r = 0; r < 8; r++) {
        float rs = 0.0f;
        int prow = r + ((lane < 16) ? 0 : 8);
        for (int j = 0; j < 4; j++) {
          float p = __expf(s[j][r] - mrow[r]);
          rs += p;
          P[wave][prow][j * 16 + l15] = f32_to_bf16(p);   // C-layout -> LDS
        }
        for (int off = 1; off < 16; off <<= 1)
          rs += __shfl_xor(rs, off, 32);
        lrow[r] = lrow[r] * alpha[r] + rs;
      }
      for (int jd = 0; jd < 8; jd++)
        for (int r = 0; r < 8; r++) o[jd][r] *= alpha[r];

      // O += P V  (P re-read in A-layout; V-frags from LDS, k split {0-7,16-23}/{8-15,24-31})
      const unsigned short* Pw = &P[wave][0][0];
      for (int kk2 = 0; kk2 < 2; kk2++) {
        FragB pa;
        const unsigned short* pp = Pw + (size_t)l15 * 72 + kk2 * 32 + aoff;
        pa.u[0] = *(const uint4v*)pp;
        pa.u[1] = *(const uint4v*)(pp + 16);
        for (int jd = 0; jd < 8; jd++) {
          const unsigned short* pv = &Vt[cur][jd * 16 + l15][kk2 * 32 + boff];
          FragB vf;
          vf.u[0] = *(const uint4v*)pv;
          vf.u[1] = *(const uint4v*)(pv + 16);
          o[jd] = __builtin_amdgcn_wmma_f32_16x16x32_bf16(
              false, pa.v, false, vf.v, (short)0, o[jd], false, false);
        }
      }
    }
    __syncthreads();                         // all reads of buf cur complete
  }

  // normalize and write ctx [B,T,C] bf16
  unsigned short* cb = ctx + ((size_t)bb * SEQ) * D_MODEL + h * HEAD_DIM;
  for (int r = 0; r < 8; r++) {
    float inv = 1.0f / lrow[r];
    int t = q0 + r + ((lane < 16) ? 0 : 8);
    for (int jd = 0; jd < 8; jd++)
      cb[(size_t)t * D_MODEL + jd * 16 + l15] = f32_to_bf16(o[jd][r] * inv);
  }
}

// ---------------------------------------------------------------------------
extern "C" void kernel_launch(void* const* d_in, const int* in_sizes, int n_in,
                              void* d_out, int out_size, void* d_ws, size_t ws_size,
                              hipStream_t stream) {
  const float* x     = (const float*)d_in[0];
  const float* w_qkv = (const float*)d_in[1];
  const float* b_qkv = (const float*)d_in[2];
  const float* w_out = (const float*)d_in[3];
  const float* b_out = (const float*)d_in[4];
  float* out = (float*)d_out;

  char* ws = (char*)d_ws;
  size_t off = 0;
  unsigned short* xbf   = (unsigned short*)(ws + off); off += (size_t)M_TOTAL * KDIM * 2;      // 32 MiB
  unsigned short* wqkvT = (unsigned short*)(ws + off); off += (size_t)N_QKV * KDIM * 2;        // 24 MiB
  unsigned short* woutT = (unsigned short*)(ws + off); off += (size_t)D_MODEL * KDIM * 2;      //  8 MiB
  unsigned short* qbuf  = (unsigned short*)(ws + off); off += (size_t)M_TOTAL * D_MODEL * 2;   // 32 MiB
  unsigned short* kbuf  = (unsigned short*)(ws + off); off += (size_t)M_TOTAL * D_MODEL * 2;   // 32 MiB
  unsigned short* vtbuf = (unsigned short*)(ws + off); off += (size_t)M_TOTAL * D_MODEL * 2;   // 32 MiB
  unsigned short* ctxbf = (unsigned short*)(ws + off); off += (size_t)M_TOTAL * D_MODEL * 2;   // 32 MiB
  (void)ws_size; (void)in_sizes; (void)n_in; (void)out_size;

  // 1) precision conversion / weight transpose
  cvt_bf16<<<(M_TOTAL * KDIM) / (256 * 4), 256, 0, stream>>>(x, xbf, M_TOTAL * KDIM);
  dim3 tb(32, 8);
  cvt_transpose_bf16<<<dim3(N_QKV / 32, KDIM / 32), tb, 0, stream>>>(w_qkv, wqkvT, KDIM, N_QKV);
  cvt_transpose_bf16<<<dim3(D_MODEL / 32, KDIM / 32), tb, 0, stream>>>(w_out, woutT, KDIM, D_MODEL);

  // 2) QKV projection (scatter to head-major q/k, transposed v)
  gemm_bf16<<<dim3(N_QKV / 128, M_TOTAL / 128), 256, 0, stream>>>(
      xbf, wqkvT, b_qkv, 0, qbuf, kbuf, vtbuf, nullptr);

  // 3) causal flash attention (async double-buffered K/V staging)
  flash_attn<<<dim3(SEQ / 128, BATCH * N_HEADS), 256, 0, stream>>>(
      qbuf, kbuf, vtbuf, ctxbf);

  // 4) output projection -> fp32 d_out
  gemm_bf16<<<dim3(D_MODEL / 128, M_TOTAL / 128), 256, 0, stream>>>(
      ctxbf, woutT, b_out, 1, nullptr, nullptr, nullptr, out);
}
